// Model_42417097015675
// MI455X (gfx1250) — matrix operand order repaired and checked by reading, not compiled, not run on hardware
//
#include <hip/hip_runtime.h>
#include <hip/hip_bf16.h>

#define SEQ   365
#define BATCH 256
#define HID   256
#define INSZ  32
#define GATES 1024
#define KTOT  288      // 32 (x) + 256 (h)
#define BT    16       // batch rows per workgroup
#define THREADS 1024   // 32 waves of 32

typedef __attribute__((ext_vector_type(16))) __bf16 v16bf;
typedef __attribute__((ext_vector_type(8)))  __bf16 v8bf;
typedef __attribute__((ext_vector_type(8)))  float  v8f;
typedef __attribute__((ext_vector_type(4)))  float  v4f;

struct bfrag { v8bf lo, hi; };

// ---------------------------------------------------------------------------
// One-time prep: pack [w_ih ; w_hh] (288 x 1024, f32) into bf16 tiles laid out
// as B fragments for v_wmma_f32_16x16x32_bf16, in contiguous k order (same as the A loads):
//   tile (kt, nt) covers K = kt*32..+31, N = nt*16..+15 (512 bf16 = 1KB)
//   within a tile: lane l owns column n0 + (l&15), K-halves 16*(l>>4)+{0..15},
//   stored as 16 consecutive bf16 per lane -> one 32B load in the main kernel.
// ---------------------------------------------------------------------------
__global__ void pack_weights(const float* __restrict__ wih,
                             const float* __restrict__ whh,
                             __bf16* __restrict__ wp) {
    int tile = blockIdx.x;            // 0 .. 9*64-1
    int lane = threadIdx.x;           // 0 .. 31
    int kt = tile >> 6, nt = tile & 63;
    int n     = nt * 16 + (lane & 15);
    int kbase = kt * 32 + ((lane >> 4) << 4);
    __bf16* dst = wp + (size_t)tile * 512 + lane * 16;
#pragma unroll
    for (int e = 0; e < 16; ++e) {    // e = 2*vgpr + pair  == k offset
        int k = kbase + e;
        float v = (k < INSZ) ? wih[(size_t)k * GATES + n]
                             : whh[(size_t)(k - INSZ) * GATES + n];
        dst[e] = (__bf16)v;
    }
}

__device__ __forceinline__ float sigf(float x)     { return 1.0f / (1.0f + __expf(-x)); }
__device__ __forceinline__ float tanhfast(float x) { return 2.0f / (1.0f + __expf(-2.0f * x)) - 1.0f; }

// ---------------------------------------------------------------------------
// Persistent-per-launch LSTM: 16 WGs, each owns 16 batch rows for all 365
// steps. Per step: gates(16x1024) = [x_t|h](16x288) @ Wpacked via bf16 WMMA.
// Weights stream from L2 each step. A scalar offset is laundered through an
// empty asm so the compiler can neither hoist the 144 B-fragment VGPRs out of
// the t-loop (scratch spill) nor lose the global address space (flat loads).
// ---------------------------------------------------------------------------
__launch_bounds__(THREADS, 1)
__global__ void lstm_kernel(const float* __restrict__ x,
                            const __bf16* __restrict__ wp,
                            const float* __restrict__ bias,
                            const float* __restrict__ fc_w,
                            const float* __restrict__ fc_b,
                            float* __restrict__ out,
                            float* __restrict__ h_out,
                            float* __restrict__ c_out) {
    constexpr int A_STRIDE = KTOT + 8;    // bf16 elems; 16B-aligned rows, bank-skewed
    constexpr int G_STRIDE = GATES + 4;   // f32 elems; bank-skewed
    __shared__ __bf16 sh_a[BT * A_STRIDE];   // [x_t(32) | h(256)] per row, bf16
    __shared__ float  sh_g[BT * G_STRIDE];   // gate staging, f32

    const int tid  = threadIdx.x;
    const int lane = tid & 31;
    const int wave = tid >> 5;               // 0..31, owns gate cols [wave*32, +32)
    const int b0   = blockIdx.x * BT;

    // elementwise mapping: thread -> (row er, 4 h-columns at en)
    const int er = tid >> 6;
    const int en = (tid & 63) << 2;

    const int arow  = lane & 15;
    const int ahalf = lane >> 4;
    const int n0a = wave * 32 + (lane & 15);
    const int n0b = n0a + 16;
    const float bias_a = bias[n0a];
    const float bias_b = bias[n0b];

    // per-wave base offset into the packed weight buffer (elements)
    const size_t wbase = (size_t)(wave * 2) * 512 + (size_t)lane * 16;

    // h := 0
    for (int i = tid; i < BT * A_STRIDE; i += THREADS) sh_a[i] = (__bf16)0.0f;
    v4f c_reg = {0.f, 0.f, 0.f, 0.f};
    __syncthreads();

    for (int t = 0; t < SEQ; ++t) {
        // Opaque scalar offset: forces B reload each step (no hoist/spill)
        // while keeping pointer provenance -> global_load_b128, not flat.
        size_t woff = 0;
        asm volatile("" : "+s"(woff));
        const __bf16* wpt = wp + wbase + woff;

        // stage x_t tile (16 x 32) into sh_a cols [0,32), bf16
        if (tid < BT * INSZ) {
            int r = tid >> 5, col = tid & 31;
            sh_a[r * A_STRIDE + col] =
                (__bf16)x[((size_t)(b0 + r) * SEQ + t) * INSZ + col];
        }
        __syncthreads();

        v8f acc0, acc1;
#pragma unroll
        for (int i = 0; i < 8; ++i) { acc0[i] = bias_a; acc1[i] = bias_b; }

#pragma unroll
        for (int kt = 0; kt < 9; ++kt) {
            const int k0 = kt * 32;
            // A fragment: two contiguous 16B LDS loads per lane (K = 16*half + 0..15, matches packed B)
            const __bf16* ap = &sh_a[arow * A_STRIDE + k0 + ahalf * 16];
            bfrag af;
            af.lo = *(const v8bf*)(ap);
            af.hi = *(const v8bf*)(ap + 8);
            v16bf a = __builtin_bit_cast(v16bf, af);

            // B fragments: pre-packed, one 32B contiguous load per lane per tile
            const v16bf* bp = (const v16bf*)(wpt + (size_t)kt * 64 * 512);
            v16bf bb0 = bp[0];
            v16bf bb1 = bp[32];   // +512 bf16 = next N-tile

            acc0 = __builtin_amdgcn_wmma_f32_16x16x32_bf16(
                       false, a, false, bb0, (short)0, acc0, false, false);
            acc1 = __builtin_amdgcn_wmma_f32_16x16x32_bf16(
                       false, a, false, bb1, (short)0, acc1, false, false);
        }

        // spill accumulators (C/D layout: vgpr i, lane l -> row i+8*(l>>4), col n0+l%16)
        {
            const int row0 = ahalf * 8;
#pragma unroll
            for (int i = 0; i < 8; ++i) {
                sh_g[(row0 + i) * G_STRIDE + n0a] = acc0[i];
                sh_g[(row0 + i) * G_STRIDE + n0b] = acc1[i];
            }
        }
        __syncthreads();

        // elementwise gate math (fp32): f,i,o,g at cols n, 256+n, 512+n, 768+n
        {
            float* gr = &sh_g[er * G_STRIDE];
            v4f f4 = *(const v4f*)(gr + en);
            v4f i4 = *(const v4f*)(gr + 256 + en);
            v4f o4 = *(const v4f*)(gr + 512 + en);
            v4f g4 = *(const v4f*)(gr + 768 + en);
            v4f h4;
#pragma unroll
            for (int e = 0; e < 4; ++e) {
                float fs = sigf(f4[e]);
                float is = sigf(i4[e]);
                float os = sigf(o4[e]);
                float gt = tanhfast(g4[e]);
                float cn = fs * c_reg[e] + is * gt;
                c_reg[e] = cn;
                h4[e] = os * tanhfast(cn);
                sh_a[er * A_STRIDE + INSZ + en + e] = (__bf16)h4[e];  // h for next step
            }
            size_t go = ((size_t)(b0 + er) * SEQ + t) * HID + en;
            *(v4f*)(h_out + go) = h4;
            *(v4f*)(c_out + go) = c_reg;
            if (t == SEQ - 1) *(v4f*)(gr + en) = h4;   // stash fp32 last_h for FC
        }
        __syncthreads();
    }

    // fused FC: out[b] = last_h[b,:] . fc_w + fc_b   (16 rows per WG, trivial)
    if (tid < BT) {
        float s = fc_b[0];
        const float* hr = &sh_g[tid * G_STRIDE];
        for (int n = 0; n < HID; ++n) s += hr[n] * fc_w[n];
        out[b0 + tid] = s;
    }
}

extern "C" void kernel_launch(void* const* d_in, const int* in_sizes, int n_in,
                              void* d_out, int out_size, void* d_ws, size_t ws_size,
                              hipStream_t stream) {
    const float* x_d  = (const float*)d_in[0];
    const float* w_ih = (const float*)d_in[1];
    const float* w_hh = (const float*)d_in[2];
    const float* bias = (const float*)d_in[3];
    const float* fc_w = (const float*)d_in[4];
    const float* fc_b = (const float*)d_in[5];

    float* out   = (float*)d_out;                       // 256
    float* h_out = out + BATCH;                         // 256*365*256
    float* c_out = h_out + (size_t)BATCH * SEQ * HID;   // 256*365*256

    __bf16* wp = (__bf16*)d_ws;                         // 288*1024 bf16 = 576 KB

    pack_weights<<<dim3(9 * 64), dim3(32), 0, stream>>>(w_ih, w_hh, wp);
    lstm_kernel<<<dim3(BATCH / BT), dim3(THREADS), 0, stream>>>(
        x_d, wp, bias, fc_w, fc_b, out, h_out, c_out);
}
